// Multihead_Attention_T_Branch_54863912239661
// MI455X (gfx1250) — matrix operand order repaired and checked by reading, not compiled, not run on hardware
//
#include <hip/hip_runtime.h>
#include <math.h>

// ---------------------------------------------------------------------------
// Types for CDNA5 WMMA (wave32, bf16 in / f32 accumulate)
// ---------------------------------------------------------------------------
typedef __bf16 bf16;
typedef __attribute__((ext_vector_type(16))) __bf16 v16bf;
typedef __attribute__((ext_vector_type(8)))  float  v8f;

#define NB 400      // B = N*F = 4*100
#define TT 200      // T
#define CC 64       // C
#define HH 16       // hidden
#define FF 100      // F
#define BT (NB*TT)  // 80000 rows
#define NL 24       // 8 branches * (q,k,v)

__device__ __forceinline__ bf16 f2bf(float f) {
  union { float f; unsigned u; } v; v.f = f;
  unsigned r = v.u + 0x7FFFu + ((v.u >> 16) & 1u);   // round-to-nearest-even
  unsigned short h = (unsigned short)(r >> 16);
  bf16 b;
  __builtin_memcpy(&b, &h, sizeof(b));
  return b;
}

__device__ __forceinline__ float bf2f(bf16 b) {
  unsigned short h;
  __builtin_memcpy(&h, &b, sizeof(h));
  union { unsigned u; float f; } v; v.u = ((unsigned)h) << 16;
  return v.f;
}

__device__ __forceinline__ float sigmoidf_(float x) {
  return 1.0f / (1.0f + expf(-x));
}

// ---------------------------------------------------------------------------
// Kernel 0: one-time weight conversion Wih f32 -> bf16 (same [l][g][c] layout;
// c is contiguous, matching the contiguous K order this GEMM uses for A and B).
// ---------------------------------------------------------------------------
__global__ void wcvt_kernel(const float* __restrict__ Wih,
                            bf16* __restrict__ Wb) {
  int idx = blockIdx.x * blockDim.x + threadIdx.x;   // 0..98303
  if (idx < NL*64*64) Wb[idx] = f2bf(Wih[idx]);
}

// ---------------------------------------------------------------------------
// Kernel 1: LayerNorm1 over C (per real/imag), output bf16 xln[2][B*T][64]
// ---------------------------------------------------------------------------
__global__ void ln1_kernel(const float* __restrict__ inp,
                           const float* __restrict__ w,
                           const float* __restrict__ bia,
                           bf16* __restrict__ xln) {
  int bt = blockIdx.x;           // 0..79999 = b*TT + t
  int c  = threadIdx.x;          // 0..63
  int b = bt / TT, t = bt % TT;
  int n = b / FF, f = b % FF;
  size_t base = ((((size_t)n*CC + c)*FF + f)*TT + t)*2;
  float xr = inp[base + 0];
  float xi = inp[base + 1];
  __shared__ float sr[CC], si[CC];
  __shared__ float stat[4];
  sr[c] = xr; si[c] = xi;
  __syncthreads();
  if (c < 2) {
    const float* s = (c == 0) ? sr : si;
    float m = 0.f, q = 0.f;
    for (int k = 0; k < CC; ++k) m += s[k];
    m /= (float)CC;
    for (int k = 0; k < CC; ++k) { float d = s[k] - m; q += d*d; }
    q /= (float)CC;
    stat[c*2+0] = m;
    stat[c*2+1] = rsqrtf(q + 1e-5f);
  }
  __syncthreads();
  float yr = (xr - stat[0]) * stat[1] * w[c] + bia[c];
  float yi = (xi - stat[2]) * stat[3] * w[c] + bia[c];
  xln[((size_t)0*BT + bt)*CC + c] = f2bf(yr);
  xln[((size_t)1*BT + bt)*CC + c] = f2bf(yi);
}

// ---------------------------------------------------------------------------
// Kernel 2: WMMA GEMM for ALL 24 LSTMs in one launch.
//   xg[l][row][g] = xln[sel(l)][row][:] . Wih[l][g][:] + bih[l][g]+bhh[l][g]
// One wave computes a 16x64 output tile (4 N-tiles x 2 K-tiles = 8 WMMAs).
// A and B operands are both pre-converted bf16 -> pure vector loads.
// ---------------------------------------------------------------------------
__global__ void xg_gemm_kernel(const bf16*  __restrict__ xln,
                               const bf16*  __restrict__ Wb,
                               const float* __restrict__ bih,
                               const float* __restrict__ bhh,
                               bf16* __restrict__ xg) {
  int rt   = blockIdx.x;                 // row tile 0..4999
  int l    = blockIdx.y;                 // 0..23
  int lane = threadIdx.x;                // 0..31
  int branch = l / 3, qkv = l % 3;
  // Q_IDX=0xCC, K_IDX=0xAA, V_IDX=0x96 (bit b = branch b's real/imag select)
  unsigned selbits = (qkv == 0) ? 0xCCu : (qkv == 1) ? 0xAAu : 0x96u;
  int sel = (selbits >> branch) & 1;

  int m   = lane & 15;
  int hiA = (lane >= 16) ? 8 : 0;        // C row-half offset
  int hiB = (lane >= 16) ? 16 : 0;       // B K-half offset
  int row = rt*16 + m;
  const bf16* aptr = xln + ((size_t)sel*BT + row)*CC;

  v8f acc[4];
  #pragma unroll
  for (int nt = 0; nt < 4; ++nt)
    #pragma unroll
    for (int j = 0; j < 8; ++j) acc[nt][j] = 0.0f;

  #pragma unroll
  for (int kt = 0; kt < 2; ++kt) {
    v16bf A;
    #pragma unroll
    for (int i = 0; i < 16; ++i) {
      int k = kt*32 + hiB + i;
      A[i] = aptr[k];
    }
    #pragma unroll
    for (int nt = 0; nt < 4; ++nt) {
      int nn = nt*16 + m;
      const bf16* wptr = Wb + (size_t)l*4096 + (size_t)nn*64 + (kt*32 + hiB);
      v16bf Bv;
      #pragma unroll
      for (int i = 0; i < 16; ++i) Bv[i] = wptr[i];
      acc[nt] = __builtin_amdgcn_wmma_f32_16x16x32_bf16(
          false, A, false, Bv, (short)0, acc[nt], false, false);
    }
  }
  #pragma unroll
  for (int nt = 0; nt < 4; ++nt) {
    int g = nt*16 + m;
    float bias = bih[(size_t)l*64 + g] + bhh[(size_t)l*64 + g];
    #pragma unroll
    for (int j = 0; j < 8; ++j) {
      int ro = rt*16 + j + hiA;
      xg[((size_t)l*BT + ro)*64 + g] = f2bf(acc[nt][j] + bias);
    }
  }
}

// ---------------------------------------------------------------------------
// Kernel 3: sequential LSTM scan, ALL 24 LSTMs at once (9600 sequences).
// Whh cached in LDS per block (one l per block), h/c in registers.
// ---------------------------------------------------------------------------
__global__ void lstm_scan_kernel(const bf16*  __restrict__ xg,
                                 const float* __restrict__ Whh,
                                 bf16* __restrict__ hbuf) {
  __shared__ float W[64*16];
  int l   = blockIdx.y;                  // 0..23
  int tid = threadIdx.x;
  for (int idx = tid; idx < 1024; idx += blockDim.x)
    W[idx] = Whh[(size_t)l*1024 + idx];
  __syncthreads();
  int b = blockIdx.x * blockDim.x + tid;
  if (b >= NB) return;

  float h[16], c[16], nh[16];
  #pragma unroll
  for (int k = 0; k < 16; ++k) { h[k] = 0.f; c[k] = 0.f; }

  for (int t = 0; t < TT; ++t) {
    const bf16* xa = xg + ((size_t)l*BT + (size_t)b*TT + t)*64;
    bf16* hb = hbuf + (((size_t)l*NB + b)*TT + t)*16;
    #pragma unroll
    for (int k = 0; k < 16; ++k) {
      float gi = bf2f(xa[k]), gf = bf2f(xa[16+k]);
      float gg = bf2f(xa[32+k]), go = bf2f(xa[48+k]);
      #pragma unroll
      for (int kk = 0; kk < 16; ++kk) {
        float hv = h[kk];
        gi += hv * W[(k     )*16 + kk];
        gf += hv * W[(16 + k)*16 + kk];
        gg += hv * W[(32 + k)*16 + kk];
        go += hv * W[(48 + k)*16 + kk];
      }
      float cn = sigmoidf_(gf)*c[k] + sigmoidf_(gi)*tanhf(gg);
      c[k]  = cn;
      nh[k] = sigmoidf_(go)*tanhf(cn);
      hb[k] = f2bf(nh[k]);
    }
    #pragma unroll
    for (int k = 0; k < 16; ++k) h[k] = nh[k];
  }
}

// ---------------------------------------------------------------------------
// Kernel 4: attention for one (branch,batch) x one 16-row tile.
// energy = QK^T/4 (13 WMMAs, K=16 padded to 32); softmax(energy*mask) in LDS
// (masked entries contribute exp(0), matching the reference); O = P.V
// (7 WMMAs, P kept in LDS as bf16 so the A operand is a plain ds load).
// ---------------------------------------------------------------------------
__global__ void attention_kernel(const bf16* __restrict__ hbuf,
                                 float* __restrict__ attb) {
  __shared__ float P[16][224];           // exp values, 208 cols + pad
  __shared__ bf16  Pb[16][224];          // normalized softmax in bf16
  int rt = blockIdx.x;                   // 0..12
  int p  = blockIdx.y;                   // 0..3199
  int branch = p / NB, b = p % NB;
  int lane = threadIdx.x;
  int m    = lane & 15;
  int hiA  = (lane >= 16) ? 8 : 0;
  bool hiHalf = (lane >= 16);

  const bf16* Q = hbuf + ((size_t)(branch*3 + 0)*NB + b) * (size_t)TT * 16;
  const bf16* K = hbuf + ((size_t)(branch*3 + 1)*NB + b) * (size_t)TT * 16;
  const bf16* V = hbuf + ((size_t)(branch*3 + 2)*NB + b) * (size_t)TT * 16;

  // A = Q row tile, K-dim 16 zero-padded to 32
  v16bf Aq;
  {
    int tq = rt*16 + m;
    #pragma unroll
    for (int i = 0; i < 16; ++i) {
      int k = (hiHalf ? 16 : 0) + i;
      Aq[i] = (k < 16 && tq < TT) ? Q[(size_t)tq*16 + k] : f2bf(0.f);
    }
  }

  for (int jt = 0; jt < 13; ++jt) {
    int s = jt*16 + m;
    v16bf Bk;
    #pragma unroll
    for (int i = 0; i < 16; ++i)         // k = hiB + i; only k<16 valid
      Bk[i] = (!hiHalf && s < TT) ? K[(size_t)s*16 + i] : f2bf(0.f);
    v8f e = {};
    e = __builtin_amdgcn_wmma_f32_16x16x32_bf16(
        false, Aq, false, Bk, (short)0, e, false, false);
    #pragma unroll
    for (int j = 0; j < 8; ++j) {
      int tg = rt*16 + j + hiA;
      float v;
      if (s >= TT)       v = -3.0e38f;             // pad column: excluded
      else if (tg >= TT) v = 0.f;                  // pad row: unused
      else               v = (s <= tg) ? e[j]*0.25f : 0.f;  // mult. mask
      P[j + hiA][s] = v;
    }
  }
  __syncthreads();
  if (lane < 16) {                       // per-row softmax over 208 cols
    float mx = -3.4e38f;
    for (int sc = 0; sc < 208; ++sc) mx = fmaxf(mx, P[lane][sc]);
    float sum = 0.f;
    for (int sc = 0; sc < 208; ++sc) {
      float ee = expf(P[lane][sc] - mx);
      P[lane][sc] = ee; sum += ee;
    }
    float inv = 1.0f / sum;
    for (int sc = 0; sc < 224; ++sc)
      Pb[lane][sc] = f2bf((sc < 208) ? P[lane][sc]*inv : 0.f);
  }
  __syncthreads();

  v8f o;
  #pragma unroll
  for (int j = 0; j < 8; ++j) o[j] = 0.0f;
  for (int kk = 0; kk < 7; ++kk) {
    v16bf Ap, Bv;
    #pragma unroll
    for (int i = 0; i < 16; ++i) {
      int kg = kk*32 + (hiHalf ? 16 : 0) + i;   // < 224
      Ap[i] = Pb[m][kg];
    }
    int kb = kk*32 + (hiHalf ? 16 : 0);
    #pragma unroll
    for (int i = 0; i < 16; ++i) {
      int sg = kb + i;
      Bv[i] = (sg < TT) ? V[(size_t)sg*16 + m] : f2bf(0.f);
    }
    o = __builtin_amdgcn_wmma_f32_16x16x32_bf16(
        false, Ap, false, Bv, (short)0, o, false, false);
  }
  #pragma unroll
  for (int j = 0; j < 8; ++j) {
    int tg = rt*16 + j + hiA;
    if (tg < TT)
      attb[(((size_t)branch*NB + b)*TT + tg)*16 + m] = o[j];
  }
}

// ---------------------------------------------------------------------------
// Kernel 5: branch combine, LN2(16), complex linear 16->64, LN3(64),
// PReLU, residual. One block per (b,t), 64 threads (one per C).
// ---------------------------------------------------------------------------
__global__ void post_kernel(const float* __restrict__ attb,
                            const float* __restrict__ inp,
                            const float* __restrict__ ln2w, const float* __restrict__ ln2b,
                            const float* __restrict__ lrw,  const float* __restrict__ lrb,
                            const float* __restrict__ liw,  const float* __restrict__ lib,
                            const float* __restrict__ ln3w, const float* __restrict__ ln3b,
                            const float* __restrict__ pa,
                            float* __restrict__ out) {
  int bt = blockIdx.x, tid = threadIdx.x;
  int b = bt / TT, t = bt % TT;
  int n = b / FF, fq = b % FF;
  __shared__ float y[2][16], yn[2][16], obuf[2][64];
  __shared__ float stat[4], stat3[4];

  if (tid < 32) {
    int f = tid & 15, ri = tid >> 4;
    float a0 = attb[(((size_t)(ri*4+0)*NB + b)*TT + t)*16 + f];
    float a1 = attb[(((size_t)(ri*4+1)*NB + b)*TT + t)*16 + f];
    float a2 = attb[(((size_t)(ri*4+2)*NB + b)*TT + t)*16 + f];
    float a3 = attb[(((size_t)(ri*4+3)*NB + b)*TT + t)*16 + f];
    y[ri][f] = (ri == 0) ? (a0 - a1 - a2 - a3) : (a0 + a1 + a2 - a3);
  }
  __syncthreads();
  if (tid < 2) {
    float mm = 0.f, q = 0.f;
    for (int f = 0; f < 16; ++f) mm += y[tid][f];
    mm /= 16.f;
    for (int f = 0; f < 16; ++f) { float d = y[tid][f] - mm; q += d*d; }
    q /= 16.f;
    stat[tid*2] = mm; stat[tid*2+1] = rsqrtf(q + 1e-5f);
  }
  __syncthreads();
  if (tid < 32) {
    int f = tid & 15, ri = tid >> 4;
    yn[ri][f] = (y[ri][f] - stat[ri*2]) * stat[ri*2+1] * ln2w[f] + ln2b[f];
  }
  __syncthreads();

  int c = tid;
  float pr_r = 0.f, pr_i = 0.f, pi_r = 0.f, pi_i = 0.f;
  #pragma unroll
  for (int f = 0; f < 16; ++f) {
    float wr = lrw[c*16 + f], wi = liw[c*16 + f];
    pr_r += yn[0][f]*wr;   // pr(x_real)
    pr_i += yn[1][f]*wr;   // pr(x_imag)
    pi_r += yn[0][f]*wi;   // pi(x_real)
    pi_i += yn[1][f]*wi;   // pi(x_imag)
  }
  obuf[0][c] = (pr_r + lrb[c]) - (pi_i + lib[c]);   // out_real
  obuf[1][c] = (pr_i + lrb[c]) + (pi_r + lib[c]);   // out_imag
  __syncthreads();
  if (tid < 2) {
    float mm = 0.f, q = 0.f;
    for (int k = 0; k < 64; ++k) mm += obuf[tid][k];
    mm /= 64.f;
    for (int k = 0; k < 64; ++k) { float d = obuf[tid][k] - mm; q += d*d; }
    q /= 64.f;
    stat3[tid*2] = mm; stat3[tid*2+1] = rsqrtf(q + 1e-5f);
  }
  __syncthreads();
  float alpha = pa[0];
  size_t ibase = ((((size_t)n*CC + c)*FF + fq)*TT + t)*2;
  #pragma unroll
  for (int ri = 0; ri < 2; ++ri) {
    float v = (obuf[ri][c] - stat3[ri*2]) * stat3[ri*2+1] * ln3w[c] + ln3b[c];
    v = (v >= 0.f) ? v : alpha*v;
    out[ibase + ri] = v + inp[ibase + ri];
  }
}

// ---------------------------------------------------------------------------
// Launch
// ---------------------------------------------------------------------------
extern "C" void kernel_launch(void* const* d_in, const int* in_sizes, int n_in,
                              void* d_out, int out_size, void* d_ws, size_t ws_size,
                              hipStream_t stream) {
  (void)in_sizes; (void)n_in; (void)out_size; (void)ws_size;
  const float* inputs = (const float*)d_in[0];
  const float* Wih    = (const float*)d_in[1];
  const float* Whh    = (const float*)d_in[2];
  const float* bih    = (const float*)d_in[3];
  const float* bhh    = (const float*)d_in[4];
  const float* ln1w   = (const float*)d_in[5];
  const float* ln1b   = (const float*)d_in[6];
  const float* ln2w   = (const float*)d_in[7];
  const float* ln2b   = (const float*)d_in[8];
  const float* lrw    = (const float*)d_in[9];
  const float* lrb    = (const float*)d_in[10];
  const float* liw    = (const float*)d_in[11];
  const float* lib    = (const float*)d_in[12];
  const float* ln3w   = (const float*)d_in[13];
  const float* ln3b   = (const float*)d_in[14];
  const float* pa     = (const float*)d_in[15];
  float* out = (float*)d_out;

  // workspace layout (bytes):
  //   xln  bf16 [2][80000][64]         @ 0           :  20,480,000
  //   hbuf bf16 [24][400][200][16]     @ 20,480,000  :  61,440,000
  //   attb f32  [8][400][200][16]      @ 81,920,000  :  40,960,000
  //   xg   bf16 [24][80000][64]        @ 122,880,000 : 245,760,000
  //   Wb   bf16 [24][64][64]           @ 368,640,000 :     196,608
  char* ws = (char*)d_ws;
  bf16*  xln  = (bf16*)(ws);
  bf16*  hbuf = (bf16*)(ws + 20480000);
  float* attb = (float*)(ws + 81920000);
  bf16*  xg   = (bf16*)(ws + 122880000);
  bf16*  Wb   = (bf16*)(ws + 368640000);

  wcvt_kernel<<<(NL*64*64 + 255)/256, 256, 0, stream>>>(Wih, Wb);
  ln1_kernel<<<BT, 64, 0, stream>>>(inputs, ln1w, ln1b, xln);
  xg_gemm_kernel<<<dim3(BT/16, NL), 32, 0, stream>>>(xln, Wb, bih, bhh, xg);
  lstm_scan_kernel<<<dim3((NB + 127)/128, NL), 128, 0, stream>>>(xg, Whh, hbuf);
  attention_kernel<<<dim3(13, 8*NB), 32, 0, stream>>>(hbuf, attb);
  post_kernel<<<BT, 64, 0, stream>>>(attb, inputs, ln2w, ln2b,
                                     lrw, lrb, liw, lib, ln3w, ln3b, pa, out);
}
